// HSCATB_82961588289848
// MI455X (gfx1250) — matrix operand order repaired and checked by reading, not compiled, not run on hardware
//
#include <hip/hip_runtime.h>

typedef __attribute__((ext_vector_type(16))) _Float16 v16h;
typedef __attribute__((ext_vector_type(8)))  float    v8f;

union Frag16 { v16h v; _Float16 h[16]; unsigned u[8]; uint4 q[2]; };
union FragF  { v8f  v; float f[8]; };

static constexpr int NTOK = 65536;            // B*H*W = 16*64*64
// ---- workspace layout (bytes), liveness-based reuse ----
static constexpr size_t OFF_XN1  = 0;                                   // f16 NTOK*256
static constexpr size_t OFF_QKV  = OFF_XN1 + (size_t)NTOK*256*2;        // f16 NTOK*768
static constexpr size_t OFF_XCAT = OFF_QKV + (size_t)NTOK*768*2;        // f16 NTOK*512
static constexpr size_t OFF_X1   = OFF_XCAT + (size_t)NTOK*512*2;       // f32 NTOK*256
static constexpr size_t OFF_XN2  = 0;            // reuse XN1+QKV (dead after attention)
static constexpr size_t OFF_X2   = OFF_XCAT;     // reuse XCAT (dead after proj)
static constexpr size_t OFF_XN3  = OFF_X1;       // reuse X1 (dead after casa)
static constexpr size_t OFF_H    = 0;            // f16 NTOK*1024, reuse XN2 region
static constexpr size_t OFF_XLOC = OFF_X1;       // f16 NTOK*1024, reuse XN3 (dead after fc1)
static constexpr size_t OFF_MEANH= OFF_XLOC + (size_t)NTOK*1024*2;      // f32 16*1024
static constexpr size_t OFF_YBAR = OFF_MEANH + 16*1024*4;               // f32 16*256
static constexpr size_t OFF_SBUF = OFF_YBAR + 16*256*4;                 // f32 16*256
static constexpr size_t OFF_GBIAS= OFF_SBUF + 16*256*4;                 // f32 16*256
static constexpr size_t OFF_PART = OFF_GBIAS + 16*256*4;                // f32 16*32*1024
// pre-transposed f16 weights [N][K]
static constexpr size_t OFF_WQKV = OFF_PART + (size_t)16*32*1024*4;     // 768x256
static constexpr size_t OFF_WPROJ= OFF_WQKV + (size_t)768*256*2;        // 256x512
static constexpr size_t OFF_WFC1 = OFF_WPROJ + (size_t)256*512*2;       // 1024x256
static constexpr size_t OFF_WFC2 = OFF_WFC1 + (size_t)1024*256*2;       // 256x1024

// ------- one-time weight transpose+convert: f32 [K,N] -> f16 [N,K] ----------
__global__ void hscatb_wt_kernel(const float* __restrict__ in, _Float16* __restrict__ out,
                                 int K, int N)
{
    int idx = blockIdx.x*256 + threadIdx.x;      // over N*K
    int n = idx / K, k = idx % K;
    out[(size_t)n*K + k] = (_Float16)in[(size_t)k*N + n];
}

// ---------------- LayerNorm (C=256, one block per token) ----------------
__global__ __launch_bounds__(256)
void hscatb_ln_kernel(const float* __restrict__ x, const float* __restrict__ g,
                      const float* __restrict__ b,
                      _Float16* __restrict__ o16, float* __restrict__ o32)
{
    __shared__ float red[256];
    const int row = blockIdx.x, c = threadIdx.x;
    float v = x[(size_t)row*256 + c];
    red[c] = v; __syncthreads();
    for (int s=128; s>0; s>>=1) { if (c<s) red[c]+=red[c+s]; __syncthreads(); }
    float mu = red[0] * (1.f/256.f);
    __syncthreads();
    float d = v - mu;
    red[c] = d*d; __syncthreads();
    for (int s=128; s>0; s>>=1) { if (c<s) red[c]+=red[c+s]; __syncthreads(); }
    float var = red[0] * (1.f/256.f);
    float y = d * rsqrtf(var + 1e-5f) * g[c] + b[c];
    if (o16) o16[(size_t)row*256 + c] = (_Float16)y;
    if (o32) o32[(size_t)row*256 + c] = y;
}

// ---------------- WMMA GEMM: f16 A [M,K] x f16 Bt [N,K] -> f32/f16 -----------
// Block tile 128x128, 8 waves as 4(M)x2(N), wave tile 32x64 = 2x4 WMMA 16x16x32.
// Both tiles are contiguous-row copies (b128 in / b128 out); LDS double-buffered.
// Epilogue options are compile-time. Assumes M%128==0, N%128==0, K%32==0.
template<bool HB, bool HPB, bool HR, bool O32, bool O16>
__global__ __launch_bounds__(256)
void hscatb_gemm_kernel(const _Float16* __restrict__ A, int lda,
                        const _Float16* __restrict__ Bt,      // [N][K] f16
                        const float* __restrict__ bias,       // [N]
                        const float* __restrict__ pbias,      // [16,N] per-batch
                        const float* __restrict__ resid,      // [M,N]
                        float* __restrict__ out_f32,
                        _Float16* __restrict__ out_f16,
                        int N, int K)
{
    __shared__ _Float16 sA[2][128*32];     // [m][k]
    __shared__ _Float16 sBt[2][128*32];    // [n][k]
    const int tid  = threadIdx.x;
    const int lane = tid & 31, wave = tid >> 5;
    const int warpM = wave >> 1, warpN = wave & 1;
    const int n0 = blockIdx.x * 128;
    const int m0 = blockIdx.y * 128;
    const int lrow = lane & 15, lhi = lane >> 4;
    const int tr = tid >> 2, tc4 = (tid & 3) * 8;   // 128 rows, 4 chunks of 8 halfs

    FragF acc[2][4];
    #pragma unroll
    for (int i=0;i<2;++i)
      #pragma unroll
      for (int j=0;j<4;++j)
        #pragma unroll
        for (int r=0;r<8;++r) acc[i][j].f[r] = 0.f;

    uint4 aQ[2], bQ[2];
    const int nk = K / 32;
    // each thread copies rows {tr, tr+64} chunk tc4 of both 128x32 tiles
    const _Float16* Ag0 = A  + (size_t)(m0 + tr)*lda + tc4;
    const _Float16* Ag1 = A  + (size_t)(m0 + tr + 64)*lda + tc4;
    const _Float16* Bg0 = Bt + (size_t)(n0 + tr)*K + tc4;
    const _Float16* Bg1 = Bt + (size_t)(n0 + tr + 64)*K + tc4;

    // ---- prologue: tile 0 ----
    aQ[0] = *(const uint4*)Ag0;  aQ[1] = *(const uint4*)Ag1;
    bQ[0] = *(const uint4*)Bg0;  bQ[1] = *(const uint4*)Bg1;
    *(uint4*)&sA[0][tr*32 + tc4]        = aQ[0];
    *(uint4*)&sA[0][(tr+64)*32 + tc4]   = aQ[1];
    *(uint4*)&sBt[0][tr*32 + tc4]       = bQ[0];
    *(uint4*)&sBt[0][(tr+64)*32 + tc4]  = bQ[1];

    for (int kt = 0; kt < nk; ++kt) {
        __syncthreads();
        const int cur = kt & 1;
        if (kt+1 < nk) {                       // global loads for next tile
            int k0 = (kt+1)*32;
            aQ[0] = *(const uint4*)(Ag0 + k0);  aQ[1] = *(const uint4*)(Ag1 + k0);
            bQ[0] = *(const uint4*)(Bg0 + k0);  bQ[1] = *(const uint4*)(Bg1 + k0);
        }
        if (kt+2 < nk) {                       // L2 prefetch for tile after next
            int k2 = (kt+2)*32;
            __builtin_prefetch(Ag0 + k2, 0, 1);
            __builtin_prefetch(Bg0 + k2, 0, 1);
        }

        Frag16 af[2], bf[4];
        #pragma unroll
        for (int mi=0; mi<2; ++mi) {
            int m = warpM*32 + mi*16 + lrow;
            int kb = lhi*8;
            af[mi].q[0] = *(const uint4*)&sA[cur][m*32 + kb];        // k = kb..kb+7
            af[mi].q[1] = *(const uint4*)&sA[cur][m*32 + 16 + kb];   // k = 16+kb..
        }
        #pragma unroll
        for (int nj=0; nj<4; ++nj) {
            int n = warpN*64 + nj*16 + lrow;
            int kb = lhi*8;
            bf[nj].q[0] = *(const uint4*)&sBt[cur][n*32 + kb];       // k = kb..kb+7
            bf[nj].q[1] = *(const uint4*)&sBt[cur][n*32 + 16 + kb];  // k = 16+kb..
        }
        #pragma unroll
        for (int mi=0; mi<2; ++mi)
          #pragma unroll
          for (int nj=0; nj<4; ++nj)
            acc[mi][nj].v = __builtin_amdgcn_wmma_f32_16x16x32_f16(
                false, af[mi].v, false, bf[nj].v, (short)0, acc[mi][nj].v, false, false);

        if (kt+1 < nk) {                       // store next tile to other buffer
            __syncthreads();
            const int nxt = cur ^ 1;
            *(uint4*)&sA[nxt][tr*32 + tc4]       = aQ[0];
            *(uint4*)&sA[nxt][(tr+64)*32 + tc4]  = aQ[1];
            *(uint4*)&sBt[nxt][tr*32 + tc4]      = bQ[0];
            *(uint4*)&sBt[nxt][(tr+64)*32 + tc4] = bQ[1];
        }
    }

    // ---- epilogue (compile-time straight-line) ----
    #pragma unroll
    for (int mi=0; mi<2; ++mi)
      #pragma unroll
      for (int nj=0; nj<4; ++nj) {
        int ncol = n0 + warpN*64 + nj*16 + lrow;
        #pragma unroll
        for (int r=0; r<8; ++r) {
            int mrow = m0 + warpM*32 + mi*16 + r + 8*lhi;
            float v = acc[mi][nj].f[r];
            if (HB)  v += bias[ncol];
            if (HPB) v += pbias[(mrow >> 12) * N + ncol];   // 4096 tokens/batch
            if (HR)  v += resid[(size_t)mrow*N + ncol];
            if (O32) out_f32[(size_t)mrow*N + ncol] = v;
            if (O16) out_f16[(size_t)mrow*N + ncol] = (_Float16)v;
        }
      }
}

// ---------------- Window attention (one block per 8x8 window) ----------------
// qkv row layout: [3, HEADS=8, d=32] (stride 768). Writes xcat [NTOK,512]:
// cols 0..255 = attention output, cols 256..511 = broadcast low-res value.
__global__ __launch_bounds__(256)
void hscatb_attn_kernel(const _Float16* __restrict__ qkv,
                        const _Float16* __restrict__ xn1,
                        const float* __restrict__ qkv_l_w,
                        _Float16* __restrict__ xcat)
{
    __shared__ unsigned sRow[64];
    __shared__ float    sAvg[256];
    __shared__ _Float16 sXl[256];
    __shared__ _Float16 sQ[64*32];          // [t][d]
    __shared__ _Float16 sK[64*32];          // [t][d]
    __shared__ _Float16 sVt[32*64];         // [d][t]  (transposed for P@V B-frags)
    __shared__ float    sS[64*64];
    __shared__ _Float16 sP[64*64];
    __shared__ _Float16 sOut[64*32];        // [t][d] per-head output staging

    const int tid = threadIdx.x, lane = tid & 31, wave = tid >> 5;
    const int lrow = lane & 15, lhi = lane >> 4;
    const int wid = blockIdx.x;
    const int bb = wid >> 6, rem = wid & 63, wh = rem >> 3, ww = rem & 7;
    if (tid < 64) {
        int i = tid >> 3, j = tid & 7;
        sRow[tid] = (unsigned)(bb*4096 + (wh*8 + i)*64 + (ww*8 + j));
    }
    __syncthreads();

    // window mean of LN'd x
    {
        float s = 0.f;
        for (int t=0; t<64; ++t) s += (float)xn1[(size_t)sRow[t]*256 + tid];
        sAvg[tid] = s * (1.f/64.f);
    }
    __syncthreads();
    // low-res branch: softmax over a single key == identity -> x_l = v_l
    {
        float a = 0.f;
        for (int k=0; k<256; ++k) a += sAvg[k] * qkv_l_w[(size_t)k*768 + 512 + tid];
        sXl[tid] = (_Float16)a;
    }
    __syncthreads();
    // coalesced broadcast of x_l to the 64 rows (8B vector stores)
    #pragma unroll
    for (int p=0; p<16; ++p) {
        int q = tid + p*256;                  // 0..4095 chunks of 4 halfs
        int t = q >> 6, cc = (q & 63) * 4;
        *(uint2*)&xcat[(size_t)sRow[t]*512 + 256 + cc] = *(const uint2*)&sXl[cc];
    }

    for (int h=0; h<8; ++h) {
        __syncthreads();
        // stage Q,K (8B vector copies) and V transposed
        #pragma unroll
        for (int p=0; p<2; ++p) {
            int q = tid + p*256;              // 0..511 chunks of 4 halfs
            int t = q >> 3, dc = (q & 7) * 4;
            size_t base = (size_t)sRow[t]*768 + h*32 + dc;
            *(uint2*)&sQ[t*32 + dc] = *(const uint2*)&qkv[base];
            *(uint2*)&sK[t*32 + dc] = *(const uint2*)&qkv[base + 256];
            uint2 vv = *(const uint2*)&qkv[base + 512];
            const _Float16* vh = (const _Float16*)&vv;
            sVt[(dc+0)*64 + t] = vh[0];
            sVt[(dc+1)*64 + t] = vh[1];
            sVt[(dc+2)*64 + t] = vh[2];
            sVt[(dc+3)*64 + t] = vh[3];
        }
        __syncthreads();

        // S = scale * Q K^T  (16 tiles of 16x16, K=32)
        for (int tt = wave; tt < 16; tt += 8) {
            int ti = tt >> 2, tj = tt & 3;
            Frag16 af, bf; FragF acc;
            #pragma unroll
            for (int r=0;r<8;++r) acc.f[r] = 0.f;
            int m = ti*16 + lrow, kb = lhi*8;
            af.q[0] = *(const uint4*)&sQ[m*32 + kb];
            af.q[1] = *(const uint4*)&sQ[m*32 + 16 + kb];
            int n = tj*16 + lrow, kb2 = lhi*8;        // B[k][n] = K[n][k]
            bf.q[0] = *(const uint4*)&sK[n*32 + kb2];
            bf.q[1] = *(const uint4*)&sK[n*32 + 16 + kb2];
            acc.v = __builtin_amdgcn_wmma_f32_16x16x32_f16(
                false, af.v, false, bf.v, (short)0, acc.v, false, false);
            #pragma unroll
            for (int r=0;r<8;++r)
                sS[(ti*16 + r + 8*lhi)*64 + tj*16 + lrow] = acc.f[r] * 0.0625f; // C^-0.5
        }
        __syncthreads();

        // row softmax (8 rows per wave, 2 logits per lane)
        for (int rr=0; rr<8; ++rr) {
            int row = wave*8 + rr;
            float v0 = sS[row*64 + lane];
            float v1 = sS[row*64 + 32 + lane];
            float mx = fmaxf(v0, v1);
            for (int o=16; o; o>>=1) mx = fmaxf(mx, __shfl_xor(mx, o, 32));
            float e0 = expf(v0 - mx), e1 = expf(v1 - mx);
            float sm = e0 + e1;
            for (int o=16; o; o>>=1) sm += __shfl_xor(sm, o, 32);
            float inv = 1.f / sm;
            sP[row*64 + lane]      = (_Float16)(e0 * inv);
            sP[row*64 + 32 + lane] = (_Float16)(e1 * inv);
        }
        __syncthreads();

        // O = P V (wave -> one 16x16 tile; K loop over 64) -> sOut
        {
            int mi = wave >> 1, nj = wave & 1;
            FragF acc;
            #pragma unroll
            for (int r=0;r<8;++r) acc.f[r] = 0.f;
            #pragma unroll
            for (int ks=0; ks<64; ks+=32) {
                Frag16 af, bf;
                int m = mi*16 + lrow, kb = lhi*8;
                af.q[0] = *(const uint4*)&sP[m*64 + ks + kb];
                af.q[1] = *(const uint4*)&sP[m*64 + ks + 16 + kb];
                int n = nj*16 + lrow, kb2 = ks + lhi*8;   // B[k][n] = V[k][n] = sVt[n][k]
                bf.q[0] = *(const uint4*)&sVt[n*64 + kb2];
                bf.q[1] = *(const uint4*)&sVt[n*64 + 16 + kb2];
                acc.v = __builtin_amdgcn_wmma_f32_16x16x32_f16(
                    false, af.v, false, bf.v, (short)0, acc.v, false, false);
            }
            #pragma unroll
            for (int r=0;r<8;++r)
                sOut[(mi*16 + r + 8*lhi)*32 + nj*16 + lrow] = (_Float16)acc.f[r];
        }
        __syncthreads();
        // coalesced O store (8B vector stores)
        #pragma unroll
        for (int p=0; p<2; ++p) {
            int q = tid + p*256;
            int t = q >> 3, dc = (q & 7) * 4;
            *(uint2*)&xcat[(size_t)sRow[t]*512 + h*32 + dc] = *(const uint2*)&sOut[t*32 + dc];
        }
    }
}

// ------- per-batch spatial mean, two stages for parallelism ----------------
template <typename T>
__global__ void hscatb_reduce_mean_p1(const T* __restrict__ in,
                                      float* __restrict__ partial, int Ccols)
{
    int blk = blockIdx.x;                 // 16 batches * 32 pixel-groups
    int b = blk >> 5, g = blk & 31;       // 128 pixels per group
    for (int c = threadIdx.x; c < Ccols; c += 256) {
        const T* p = in + ((size_t)b*4096 + g*128)*Ccols + c;
        float s = 0.f;
        for (int i=0; i<128; ++i) s += (float)p[(size_t)i*Ccols];
        partial[((size_t)b*32 + g)*Ccols + c] = s;
    }
}
__global__ void hscatb_reduce_mean_p2(const float* __restrict__ partial,
                                      float* __restrict__ out, int Ccols)
{
    int idx = blockIdx.x*256 + threadIdx.x;
    if (idx >= 16*Ccols) return;
    int b = idx / Ccols, c = idx % Ccols;
    float s = 0.f;
    #pragma unroll
    for (int g=0; g<32; ++g) s += partial[((size_t)b*32 + g)*Ccols + c];
    out[idx] = s * (1.f/4096.f);
}

// ---------------- SE (casa) gate: sigmoid(relu(y@w1)@w2) -------------------
__global__ __launch_bounds__(256)
void hscatb_se_kernel(const float* __restrict__ ybar, const float* __restrict__ w1,
                      const float* __restrict__ w2, float* __restrict__ sbuf)
{
    __shared__ float t16[16];
    const int b = blockIdx.x, c = threadIdx.x;
    if (c < 16) {
        float a = 0.f;
        for (int k=0; k<256; ++k) a += ybar[b*256 + k] * w1[k*16 + c];
        t16[c] = fmaxf(a, 0.f);
    }
    __syncthreads();
    float a = 0.f;
    #pragma unroll
    for (int j=0; j<16; ++j) a += t16[j] * w2[j*256 + c];
    sbuf[b*256 + c] = 1.f / (1.f + expf(-a));
}

__global__ void hscatb_casa_apply_kernel(const float* __restrict__ x1,
                                         const float* __restrict__ xn2,
                                         const float* __restrict__ sbuf,
                                         float* __restrict__ x2)
{
    size_t idx = (size_t)blockIdx.x*256 + threadIdx.x;   // NTOK*256
    int c = (int)(idx & 255);
    int b = (int)(idx >> 20);                            // 4096*256 = 2^20
    x2[idx] = x1[idx] + xn2[idx] * sbuf[b*256 + c];
}

// ---------------- depthwise 3x3 / 5x5 + exact GELU -> f16 x_local ----------
__global__ void hscatb_dwconv_gelu_kernel(const _Float16* __restrict__ hbuf,
    const float* __restrict__ w3, const float* __restrict__ b3,
    const float* __restrict__ w5, const float* __restrict__ b5,
    _Float16* __restrict__ xloc)
{
    size_t idx = (size_t)blockIdx.x*256 + threadIdx.x;   // NTOK*1024
    int ch = (int)(idx & 1023);
    size_t tok = idx >> 10;
    int b = (int)(tok >> 12);
    int pix = (int)(tok & 4095);
    int y = pix >> 6, x = pix & 63;
    size_t base = (size_t)b*4096;
    float acc;
    if (ch < 512) {
        acc = b3[ch];
        const float* w = w3 + ch*9;
        #pragma unroll
        for (int dy=-1; dy<=1; ++dy)
          #pragma unroll
          for (int dx=-1; dx<=1; ++dx) {
            int yy = y+dy, xx = x+dx;
            if (yy>=0 && yy<64 && xx>=0 && xx<64)
                acc += w[(dy+1)*3 + (dx+1)] * (float)hbuf[(base + yy*64 + xx)*1024 + ch];
          }
    } else {
        int c5 = ch - 512;
        acc = b5[c5];
        const float* w = w5 + c5*25;
        #pragma unroll
        for (int dy=-2; dy<=2; ++dy)
          #pragma unroll
          for (int dx=-2; dx<=2; ++dx) {
            int yy = y+dy, xx = x+dx;
            if (yy>=0 && yy<64 && xx>=0 && xx<64)
                acc += w[(dy+2)*5 + (dx+2)] * (float)hbuf[(base + yy*64 + xx)*1024 + ch];
          }
    }
    float g = 0.5f * acc * (1.f + erff(acc * 0.70710678118f));
    xloc[idx] = (_Float16)g;
}

// ---- fold global branch into per-batch bias: gbias[b] = fc2_b + meanh[b]@W[HID:] ----
__global__ __launch_bounds__(256)
void hscatb_gbias_kernel(const float* __restrict__ meanh, const float* __restrict__ fc2_w,
                         const float* __restrict__ fc2_b, float* __restrict__ gbias)
{
    const int b = blockIdx.x, c = threadIdx.x;
    float a = fc2_b[c];
    for (int j=0; j<1024; ++j)
        a += meanh[b*1024 + j] * fc2_w[(size_t)(1024 + j)*256 + c];
    gbias[b*256 + c] = a;
}

extern "C" void kernel_launch(void* const* d_in, const int* in_sizes, int n_in,
                              void* d_out, int out_size, void* d_ws, size_t ws_size,
                              hipStream_t stream)
{
    (void)in_sizes; (void)n_in; (void)out_size; (void)ws_size;
    const float* x       = (const float*)d_in[0];
    const float* n1g     = (const float*)d_in[1];
    const float* n1b     = (const float*)d_in[2];
    const float* qkv_h_w = (const float*)d_in[3];
    const float* qkv_l_w = (const float*)d_in[4];
    const float* proj_w  = (const float*)d_in[5];
    const float* proj_b  = (const float*)d_in[6];
    const float* n2g     = (const float*)d_in[7];
    const float* n2b     = (const float*)d_in[8];
    const float* se_w1   = (const float*)d_in[9];
    const float* se_w2   = (const float*)d_in[10];
    const float* n3g     = (const float*)d_in[11];
    const float* n3b     = (const float*)d_in[12];
    const float* fc1_w   = (const float*)d_in[13];
    const float* fc1_b   = (const float*)d_in[14];
    const float* dw3_w   = (const float*)d_in[15];
    const float* dw3_b   = (const float*)d_in[16];
    const float* dw5_w   = (const float*)d_in[17];
    const float* dw5_b   = (const float*)d_in[18];
    const float* fc2_w   = (const float*)d_in[19];
    const float* fc2_b   = (const float*)d_in[20];
    float* outp = (float*)d_out;

    char* ws = (char*)d_ws;
    _Float16* xn1  = (_Float16*)(ws + OFF_XN1);
    _Float16* qkvh = (_Float16*)(ws + OFF_QKV);
    _Float16* xcat = (_Float16*)(ws + OFF_XCAT);
    float*    x1   = (float*)   (ws + OFF_X1);
    float*    xn2  = (float*)   (ws + OFF_XN2);
    float*    x2   = (float*)   (ws + OFF_X2);
    _Float16* xn3  = (_Float16*)(ws + OFF_XN3);
    _Float16* hbuf = (_Float16*)(ws + OFF_H);
    _Float16* xloc = (_Float16*)(ws + OFF_XLOC);
    float*    meanh= (float*)   (ws + OFF_MEANH);
    float*    ybar = (float*)   (ws + OFF_YBAR);
    float*    sbuf = (float*)   (ws + OFF_SBUF);
    float*    gbias= (float*)   (ws + OFF_GBIAS);
    float*    part = (float*)   (ws + OFF_PART);
    _Float16* qkvT = (_Float16*)(ws + OFF_WQKV);
    _Float16* projT= (_Float16*)(ws + OFF_WPROJ);
    _Float16* fc1T = (_Float16*)(ws + OFF_WFC1);
    _Float16* fc2T = (_Float16*)(ws + OFF_WFC2);

    // ---- one-time weight transpose/convert to f16 [N][K] ----
    hscatb_wt_kernel<<<(768*256)/256, 256, 0, stream>>>(qkv_h_w, qkvT, 256, 768);
    hscatb_wt_kernel<<<(256*512)/256, 256, 0, stream>>>(proj_w, projT, 512, 256);
    hscatb_wt_kernel<<<(1024*256)/256, 256, 0, stream>>>(fc1_w, fc1T, 256, 1024);
    hscatb_wt_kernel<<<(256*1024)/256, 256, 0, stream>>>(fc2_w, fc2T, 1024, 256); // rows 0..1023

    // ---- block 1: x += spatial_attn(LN1(x)) ----
    hscatb_ln_kernel<<<NTOK, 256, 0, stream>>>(x, n1g, n1b, xn1, nullptr);
    hscatb_gemm_kernel<false,false,false,false,true>
        <<<dim3(768/128, NTOK/128), 256, 0, stream>>>(
        xn1, 256, qkvT, nullptr, nullptr, nullptr, nullptr, qkvh, 768, 256);
    hscatb_attn_kernel<<<1024, 256, 0, stream>>>(qkvh, xn1, qkv_l_w, xcat);
    hscatb_gemm_kernel<true,false,true,true,false>
        <<<dim3(256/128, NTOK/128), 256, 0, stream>>>(
        xcat, 512, projT, proj_b, nullptr, x, x1, nullptr, 256, 512);

    // ---- block 2: x += casa(LN2(x)) ----
    hscatb_ln_kernel<<<NTOK, 256, 0, stream>>>(x1, n2g, n2b, nullptr, xn2);
    hscatb_reduce_mean_p1<float><<<16*32, 256, 0, stream>>>(xn2, part, 256);
    hscatb_reduce_mean_p2<<<(16*256)/256, 256, 0, stream>>>(part, ybar, 256);
    hscatb_se_kernel<<<16, 256, 0, stream>>>(ybar, se_w1, se_w2, sbuf);
    hscatb_casa_apply_kernel<<<(NTOK*256)/256, 256, 0, stream>>>(x1, xn2, sbuf, x2);

    // ---- block 3: x += mlgffn(LN3(x)) ----
    hscatb_ln_kernel<<<NTOK, 256, 0, stream>>>(x2, n3g, n3b, xn3, nullptr);
    hscatb_gemm_kernel<true,false,false,false,true>
        <<<dim3(1024/128, NTOK/128), 256, 0, stream>>>(
        xn3, 256, fc1T, fc1_b, nullptr, nullptr, nullptr, hbuf, 1024, 256);
    hscatb_reduce_mean_p1<_Float16><<<16*32, 256, 0, stream>>>(hbuf, part, 1024);
    hscatb_reduce_mean_p2<<<(16*1024)/256, 256, 0, stream>>>(part, meanh, 1024);
    hscatb_dwconv_gelu_kernel<<<(NTOK*1024)/256, 256, 0, stream>>>(
        hbuf, dw3_w, dw3_b, dw5_w, dw5_b, xloc);
    hscatb_gbias_kernel<<<16, 256, 0, stream>>>(meanh, fc2_w, fc2_b, gbias);
    hscatb_gemm_kernel<false,true,true,true,false>
        <<<dim3(256/128, NTOK/128), 256, 0, stream>>>(
        xloc, 1024, fc2T, nullptr, gbias, x2, outp, nullptr, 256, 1024);
}
